// Attention_28372553957894
// MI455X (gfx1250) — matrix operand order repaired and checked by reading, not compiled, not run on hardware
//
#include <hip/hip_runtime.h>

// ---------------------------------------------------------------------------
// Problem constants (from the reference): B=8, N=2048, CIN=H=UNITS=256
// ---------------------------------------------------------------------------
#define BB   8
#define NN   2048
#define CIN  256
#define HH   256
#define UU   256

// LDS tile pitch: 40 bf16 = 80 bytes (multiple of 16B -> aligned b128 stores,
// spreads transpose-staging stores across banks).
#define LDT  40

// Block geometry: 256 threads = 8 wave32s in a 2x4 grid.
// Block tile: M=64 x N=256, K-step 32. Wave tile: 32x64 (acc[2][4]).
#define BM   64
#define BN   256

typedef __attribute__((ext_vector_type(16))) __bf16 v16bf;
typedef __attribute__((ext_vector_type(8)))  float  v8f;
typedef __attribute__((ext_vector_type(4)))  float  f32x4;
typedef __attribute__((ext_vector_type(4)))  __bf16 bf16x4;
typedef __attribute__((ext_vector_type(4)))  unsigned int u32x4;

// ---------------------------------------------------------------------------
// WMMA helpers — CDNA5 wave32, V_WMMA_F32_16X16X32_BF16
// Fragment layouts per CDNA5 ISA 7.12.2 (wave32, 16-bit data):
//   A (16x32, MxK): lane -> m=lane%16, hi=lane/16;
//                   VGPR v -> k = (v/4)*16 + hi*8 + (v%4)*2  (pair k,k+1)
//   B (32x16, KxN): lane -> n=lane%16, hi=lane/16;
//                   VGPR v -> k = (v/4)*16 + hi*8 + (v%4)*2  (pair k,k+1)
//   C/D (16x16 f32): VGPR r -> row = r + 8*(lane/16), col = lane%16
// ---------------------------------------------------------------------------
__device__ __forceinline__ v16bf load_a_frag(const __bf16* p) {
  const int lane = threadIdx.x & 31;
  const int m  = lane & 15;
  const int hi = lane >> 4;
  v16bf f;
#pragma unroll
  for (int v = 0; v < 8; ++v) {
    const int k = ((v >> 2) << 4) + (hi << 3) + ((v & 3) << 1);
    f[2 * v]     = p[m * LDT + k];
    f[2 * v + 1] = p[m * LDT + k + 1];
  }
  return f;
}

__device__ __forceinline__ v16bf load_b_frag(const __bf16* p) {
  const int lane = threadIdx.x & 31;
  const int n  = lane & 15;
  const int hi = lane >> 4;
  v16bf f;
#pragma unroll
  for (int v = 0; v < 8; ++v) {
    const int k = ((v >> 2) << 4) + (hi << 3) + ((v & 3) << 1);
    f[2 * v]     = p[n * LDT + k];
    f[2 * v + 1] = p[n * LDT + k + 1];
  }
  return f;
}

__device__ __forceinline__ v8f wmma_bf16(v16bf a, v16bf b, v8f c) {
  return __builtin_amdgcn_wmma_f32_16x16x32_bf16(false, a, false, b,
                                                 (short)0, c, false, false);
}

// 8 waves (2x4); wave tile 32x64: 2 A-frags x 4 B-frags -> 8 WMMAs per K-step.
__device__ __forceinline__ void mma_block(const __bf16* As, const __bf16* Bs,
                                          v8f (&acc)[2][4]) {
  const int wave = threadIdx.x >> 5;
  const int wm = (wave >> 2) * 32;
  const int wn = (wave & 3) * 64;
  v16bf a0 = load_a_frag(As + wm * LDT);
  v16bf a1 = load_a_frag(As + (wm + 16) * LDT);
  v16bf b[4];
#pragma unroll
  for (int j = 0; j < 4; ++j) b[j] = load_b_frag(Bs + (wn + 16 * j) * LDT);
#pragma unroll
  for (int j = 0; j < 4; ++j) acc[0][j] = wmma_bf16(a0, b[j], acc[0][j]);
#pragma unroll
  for (int j = 0; j < 4; ++j) acc[1][j] = wmma_bf16(a1, b[j], acc[1][j]);
}

// ---------------------------------------------------------------------------
// Vectorized LDS staging (256 threads; tile = ROWS/COLS x 32 K, pitch LDT)
// ---------------------------------------------------------------------------
// dst[r][k] = cvt(src[(row0+r)*ld + k0+k])  — b128 loads, b64 LDS stores
template <int ROWS>
__device__ __forceinline__ void stage_f32_rows(__bf16* dst, const float* src,
                                               int ld, int row0, int k0) {
#pragma unroll
  for (int i = 0; i < ROWS / 32; ++i) {
    const int idx4 = threadIdx.x + i * 256;
    const int r  = idx4 >> 3;               // 8 quads per row
    const int ko = (idx4 & 7) * 4;
    f32x4 v = *(const f32x4*)(src + (size_t)(row0 + r) * ld + (k0 + ko));
    bf16x4 o;
    o[0] = (__bf16)v[0]; o[1] = (__bf16)v[1];
    o[2] = (__bf16)v[2]; o[3] = (__bf16)v[3];
    *(bf16x4*)(dst + r * LDT + ko) = o;
  }
}

// dst[r][k] = src[(row0+r)*ld + k0+k]  — b128 loads, b128 LDS stores
template <int ROWS>
__device__ __forceinline__ void stage_bf16_rows(__bf16* dst, const __bf16* src,
                                                int ld, int row0, int k0) {
#pragma unroll
  for (int i = 0; i < ROWS / 64; ++i) {
    const int idx8 = threadIdx.x + i * 256;
    const int r  = idx8 >> 2;               // 4 octs per row
    const int ko = (idx8 & 3) * 8;
    u32x4 v = *(const u32x4*)(src + (size_t)(row0 + r) * ld + (k0 + ko));
    *(u32x4*)(dst + r * LDT + ko) = v;
  }
}

// dst[c][k] = cvt(src[(k0+k)*ld + c0+c])  — coalesced b128 loads over c
template <int COLS>
__device__ __forceinline__ void stage_f32_T(__bf16* dst, const float* src,
                                            int ld, int k0, int c0) {
#pragma unroll
  for (int i = 0; i < COLS / 32; ++i) {
    const int idx4 = threadIdx.x + i * 256;
    const int c4 = (idx4 & (COLS / 4 - 1)) * 4;
    const int k  = idx4 / (COLS / 4);
    f32x4 v = *(const f32x4*)(src + (size_t)(k0 + k) * ld + (c0 + c4));
#pragma unroll
    for (int j = 0; j < 4; ++j) dst[(c4 + j) * LDT + k] = (__bf16)v[j];
  }
}

// dst[c][k] = src[(k0+k)*ld + c0+c]  — coalesced b128 loads over c
template <int COLS>
__device__ __forceinline__ void stage_bf16_T(__bf16* dst, const __bf16* src,
                                             int ld, int k0, int c0) {
#pragma unroll
  for (int i = 0; i < COLS / 64; ++i) {
    const int idx8 = threadIdx.x + i * 256;
    const int c8 = (idx8 & (COLS / 8 - 1)) * 8;
    const int k  = idx8 / (COLS / 8);
    u32x4 raw = *(const u32x4*)(src + (size_t)(k0 + k) * ld + (c0 + c8));
    const __bf16* t = (const __bf16*)&raw;
#pragma unroll
    for (int j = 0; j < 8; ++j) dst[(c8 + j) * LDT + k] = t[j];
  }
}

// dst[c][k] = bf16(exp(S[(k0+k)*ld + c0+c] - rowmax)*rowinv), COLS=64 (fused P)
__device__ __forceinline__ void stage_exp_T(__bf16* dst, const float* S, int ld,
                                            int k0, int c0,
                                            const float* __restrict__ rowmax,
                                            const float* __restrict__ rowinv,
                                            int rowbase) {
#pragma unroll
  for (int i = 0; i < 2; ++i) {
    const int idx4 = threadIdx.x + i * 256;
    const int c4 = (idx4 & 15) * 4;
    const int k  = idx4 >> 4;
    const float m   = rowmax[rowbase + k0 + k];
    const float inv = rowinv[rowbase + k0 + k];
    f32x4 v = *(const f32x4*)(S + (size_t)(k0 + k) * ld + (c0 + c4));
#pragma unroll
    for (int j = 0; j < 4; ++j)
      dst[(c4 + j) * LDT + k] = (__bf16)(__expf(v[j] - m) * inv);
  }
}

// ---------------------------------------------------------------------------
// Epilogue index helper: wave tile 32x64 inside 64x256 block tile.
// ---------------------------------------------------------------------------
struct TilePos {
  int wm, wn, hi, col;
};
__device__ __forceinline__ TilePos tile_pos() {
  const int lane = threadIdx.x & 31;
  const int wave = threadIdx.x >> 5;
  TilePos t;
  t.wm  = (wave >> 2) * 32;
  t.wn  = (wave & 3) * 64;
  t.hi  = lane >> 4;
  t.col = lane & 15;
  return t;
}

// ---------------------------------------------------------------------------
// Kernel 1: q = relu(x@Wq+bq), k = relu(x@Wk+bk)   (blockIdx.z selects q/k)
// Block tile: 64 rows of B*N x all 256 H columns.
// ---------------------------------------------------------------------------
__global__ __launch_bounds__(256) void proj_kernel(
    const float* __restrict__ x, const float* __restrict__ Wq,
    const float* __restrict__ bq, const float* __restrict__ Wk,
    const float* __restrict__ bk, __bf16* __restrict__ qws,
    __bf16* __restrict__ kws) {
  const float* W    = blockIdx.z ? Wk : Wq;
  const float* bias = blockIdx.z ? bk : bq;
  __bf16*      out  = blockIdx.z ? kws : qws;

  __shared__ __bf16 As[2][BM * LDT];
  __shared__ __bf16 Bs[2][BN * LDT];
  const int m0 = blockIdx.x * BM;  // over B*N

  const v8f z = {0.f, 0.f, 0.f, 0.f, 0.f, 0.f, 0.f, 0.f};
  v8f acc[2][4] = {{z, z, z, z}, {z, z, z, z}};

  stage_f32_rows<BM>(As[0], x, CIN, m0, 0);
  stage_f32_T<BN>(Bs[0], W, HH, 0, 0);
  __syncthreads();
  for (int k0 = 0; k0 < CIN; k0 += 32) {
    const int cur = (k0 >> 5) & 1;
    if (k0 + 32 < CIN) {
      stage_f32_rows<BM>(As[cur ^ 1], x, CIN, m0, k0 + 32);
      stage_f32_T<BN>(Bs[cur ^ 1], W, HH, k0 + 32, 0);
    }
    mma_block(As[cur], Bs[cur], acc);
    __syncthreads();
  }

  const TilePos t = tile_pos();
#pragma unroll
  for (int i = 0; i < 2; ++i)
#pragma unroll
    for (int j = 0; j < 4; ++j)
#pragma unroll
      for (int r = 0; r < 8; ++r) {
        const int row = m0 + t.wm + i * 16 + r + t.hi * 8;
        const int c   = t.wn + j * 16 + t.col;
        float v = acc[i][j][r] + bias[c];
        out[(size_t)row * HH + c] = (__bf16)fmaxf(v, 0.f);
      }
}

// ---------------------------------------------------------------------------
// Kernel 2: S[b,n,m] = sum_h k[b,n,h] * q[b,m,h]   (f32 logits)
// Block tile: 64 n-rows x 256 m-cols.
// ---------------------------------------------------------------------------
__global__ __launch_bounds__(256) void scores_kernel(
    const __bf16* __restrict__ kws, const __bf16* __restrict__ qws,
    float* __restrict__ S) {
  const int b  = blockIdx.z;
  const int n0 = blockIdx.x * BM;
  const int m0 = blockIdx.y * BN;
  const __bf16* kb = kws + (size_t)b * NN * HH;
  const __bf16* qb = qws + (size_t)b * NN * HH;
  float* Sb = S + (size_t)b * NN * NN;

  __shared__ __bf16 As[2][BM * LDT];
  __shared__ __bf16 Bs[2][BN * LDT];
  const v8f z = {0.f, 0.f, 0.f, 0.f, 0.f, 0.f, 0.f, 0.f};
  v8f acc[2][4] = {{z, z, z, z}, {z, z, z, z}};

  stage_bf16_rows<BM>(As[0], kb, HH, n0, 0);
  stage_bf16_rows<BN>(Bs[0], qb, HH, m0, 0);
  __syncthreads();
  for (int k0 = 0; k0 < HH; k0 += 32) {
    const int cur = (k0 >> 5) & 1;
    if (k0 + 32 < HH) {
      stage_bf16_rows<BM>(As[cur ^ 1], kb, HH, n0, k0 + 32);
      stage_bf16_rows<BN>(Bs[cur ^ 1], qb, HH, m0, k0 + 32);
    }
    if (k0 + 64 < HH) {  // prefetch K+2 tile (global_prefetch_b8)
      __builtin_prefetch(kb + (size_t)(n0 + (threadIdx.x & 63)) * HH + k0 + 64, 0, 0);
      __builtin_prefetch(qb + (size_t)(m0 + (threadIdx.x & 255)) * HH + k0 + 64, 0, 0);
    }
    mma_block(As[cur], Bs[cur], acc);
    __syncthreads();
  }

  const TilePos t = tile_pos();
#pragma unroll
  for (int i = 0; i < 2; ++i)
#pragma unroll
    for (int j = 0; j < 4; ++j)
#pragma unroll
      for (int r = 0; r < 8; ++r) {
        const int row = n0 + t.wm + i * 16 + r + t.hi * 8;
        const int c   = m0 + t.wn + j * 16 + t.col;
        Sb[(size_t)row * NN + c] = acc[i][j][r];
      }
}

// ---------------------------------------------------------------------------
// Kernel 3: per-row softmax statistics: rowmax[b*N+n], rowinv = 1/sum(exp(.-max))
// ---------------------------------------------------------------------------
__global__ __launch_bounds__(256) void rowstats_kernel(
    const float* __restrict__ S, float* __restrict__ rowmax,
    float* __restrict__ rowinv) {
  const float* p = S + (size_t)blockIdx.x * NN;
  __shared__ float red[256];
  const int t = threadIdx.x;

  float m = -3.402823466e38f;
  for (int i = t; i < NN / 4; i += 256) {
    f32x4 v = ((const f32x4*)p)[i];
    m = fmaxf(m, fmaxf(fmaxf(v[0], v[1]), fmaxf(v[2], v[3])));
  }
  red[t] = m;
  __syncthreads();
  for (int s = 128; s > 0; s >>= 1) {
    if (t < s) red[t] = fmaxf(red[t], red[t + s]);
    __syncthreads();
  }
  m = red[0];
  __syncthreads();

  float sum = 0.f;
  for (int i = t; i < NN / 4; i += 256) {
    f32x4 v = ((const f32x4*)p)[i];
    sum += __expf(v[0] - m) + __expf(v[1] - m) +
           __expf(v[2] - m) + __expf(v[3] - m);
  }
  red[t] = sum;
  __syncthreads();
  for (int s = 128; s > 0; s >>= 1) {
    if (t < s) red[t] += red[t + s];
    __syncthreads();
  }
  if (t == 0) {
    rowmax[blockIdx.x] = m;
    rowinv[blockIdx.x] = 1.0f / red[0];
  }
}

// ---------------------------------------------------------------------------
// Kernel 4: att[b,m,h] = sum_n P[b,n,m] * q[b,n,h],  P = softmax(S) fused into
// the A-tile staging.  Block tile: 64 m-rows x all 256 h-cols -> S read once.
// ---------------------------------------------------------------------------
__global__ __launch_bounds__(256) void att_kernel(
    const float* __restrict__ S, const __bf16* __restrict__ qws,
    const float* __restrict__ rowmax, const float* __restrict__ rowinv,
    __bf16* __restrict__ attws) {
  const int b  = blockIdx.z;
  const int m0 = blockIdx.x * BM;  // over N (m)
  const float*  Sb = S   + (size_t)b * NN * NN;
  const __bf16* qb = qws + (size_t)b * NN * HH;
  const int rowbase = b * NN;

  __shared__ __bf16 As[2][BM * LDT];
  __shared__ __bf16 Bs[2][BN * LDT];
  const v8f z = {0.f, 0.f, 0.f, 0.f, 0.f, 0.f, 0.f, 0.f};
  v8f acc[2][4] = {{z, z, z, z}, {z, z, z, z}};

  stage_exp_T(As[0], Sb, NN, 0, m0, rowmax, rowinv, rowbase);
  stage_bf16_T<BN>(Bs[0], qb, HH, 0, 0);
  __syncthreads();
  for (int k0 = 0; k0 < NN; k0 += 32) {  // contraction over n
    const int cur = (k0 >> 5) & 1;
    if (k0 + 32 < NN) {
      stage_exp_T(As[cur ^ 1], Sb, NN, k0 + 32, m0, rowmax, rowinv, rowbase);
      stage_bf16_T<BN>(Bs[cur ^ 1], qb, HH, k0 + 32, 0);
    }
    mma_block(As[cur], Bs[cur], acc);
    __syncthreads();
  }

  const TilePos t = tile_pos();
#pragma unroll
  for (int i = 0; i < 2; ++i)
#pragma unroll
    for (int j = 0; j < 4; ++j)
#pragma unroll
      for (int r = 0; r < 8; ++r) {
        const int row = m0 + t.wm + i * 16 + r + t.hi * 8;
        const int c   = t.wn + j * 16 + t.col;
        attws[((size_t)b * NN + row) * HH + c] = (__bf16)acc[i][j][r];
      }
}

// ---------------------------------------------------------------------------
// Kernel 5: out = relu(att @ Wm + bm)  -> f32 d_out, row-major (B*N, UNITS)
// ---------------------------------------------------------------------------
__global__ __launch_bounds__(256) void final_kernel(
    const __bf16* __restrict__ attws, const float* __restrict__ Wm,
    const float* __restrict__ bm, float* __restrict__ out) {
  __shared__ __bf16 As[2][BM * LDT];
  __shared__ __bf16 Bs[2][BN * LDT];
  const int m0 = blockIdx.x * BM;  // over B*N

  const v8f z = {0.f, 0.f, 0.f, 0.f, 0.f, 0.f, 0.f, 0.f};
  v8f acc[2][4] = {{z, z, z, z}, {z, z, z, z}};

  stage_bf16_rows<BM>(As[0], attws, HH, m0, 0);
  stage_f32_T<BN>(Bs[0], Wm, UU, 0, 0);
  __syncthreads();
  for (int k0 = 0; k0 < HH; k0 += 32) {
    const int cur = (k0 >> 5) & 1;
    if (k0 + 32 < HH) {
      stage_bf16_rows<BM>(As[cur ^ 1], attws, HH, m0, k0 + 32);
      stage_f32_T<BN>(Bs[cur ^ 1], Wm, UU, k0 + 32, 0);
    }
    mma_block(As[cur], Bs[cur], acc);
    __syncthreads();
  }

  const TilePos t = tile_pos();
#pragma unroll
  for (int i = 0; i < 2; ++i)
#pragma unroll
    for (int j = 0; j < 4; ++j)
#pragma unroll
      for (int r = 0; r < 8; ++r) {
        const int row = m0 + t.wm + i * 16 + r + t.hi * 8;
        const int c   = t.wn + j * 16 + t.col;
        out[(size_t)row * UU + c] = fmaxf(acc[i][j][r] + bm[c], 0.f);
      }
}

// ---------------------------------------------------------------------------
// Host-side launcher. Inputs: x, Wq, bq, Wk, bk, Wm, bm (all f32).
// Workspace: q(bf16) | k(bf16) | att(bf16) | rowmax | rowinv | S(f32) ~159.5MB
// ---------------------------------------------------------------------------
extern "C" void kernel_launch(void* const* d_in, const int* in_sizes, int n_in,
                              void* d_out, int out_size, void* d_ws,
                              size_t ws_size, hipStream_t stream) {
  const float* x  = (const float*)d_in[0];
  const float* Wq = (const float*)d_in[1];
  const float* bq = (const float*)d_in[2];
  const float* Wk = (const float*)d_in[3];
  const float* bk = (const float*)d_in[4];
  const float* Wm = (const float*)d_in[5];
  const float* bm = (const float*)d_in[6];
  float* out = (float*)d_out;

  char* ws = (char*)d_ws;
  const size_t sz_proj  = (size_t)BB * NN * HH * sizeof(__bf16);  // 8 MiB each
  const size_t sz_stats = (size_t)BB * NN * sizeof(float);        // 64 KiB each
  __bf16* qws    = (__bf16*)(ws);
  __bf16* kws    = (__bf16*)(ws + sz_proj);
  __bf16* attws  = (__bf16*)(ws + 2 * sz_proj);
  float*  rowmax = (float*)(ws + 3 * sz_proj);
  float*  rowinv = (float*)(ws + 3 * sz_proj + sz_stats);
  float*  Sws    = (float*)(ws + 3 * sz_proj + 2 * sz_stats);     // 128 MiB

  // 1) q/k projections with fused bias+relu, emit bf16
  proj_kernel<<<dim3((BB * NN) / BM, 1, 2), 256, 0, stream>>>(
      x, Wq, bq, Wk, bk, qws, kws);

  // 2) logits S = k @ q^T per batch
  scores_kernel<<<dim3(NN / BM, NN / BN, BB), 256, 0, stream>>>(kws, qws, Sws);

  // 3) softmax row statistics (max, 1/sum)
  rowstats_kernel<<<dim3(BB * NN), 256, 0, stream>>>(Sws, rowmax, rowinv);

  // 4) att[b,m,h] = sum_n softmax(S)[b,n,m] q[b,n,h]  (exp fused in staging)
  att_kernel<<<dim3(NN / BM, 1, BB), 256, 0, stream>>>(
      Sws, qws, rowmax, rowinv, attws);

  // 5) out = relu(att @ Wm + bm)
  final_kernel<<<dim3((BB * NN) / BM, 1, 1), 256, 0, stream>>>(
      attws, Wm, bm, out);
}
